// CustomGRUCell_26508538150926
// MI455X (gfx1250) — matrix operand order repaired and checked by reading, not compiled, not run on hardware
//
#include <hip/hip_runtime.h>

typedef __attribute__((ext_vector_type(16))) __bf16 v16bf;
typedef __attribute__((ext_vector_type(8)))  __bf16 v8bf;
typedef __attribute__((ext_vector_type(8)))  float  v8f;

#define UNITS     128
#define BQ        16          // batch rows per block (one WMMA M-tile)
#define AS        136         // padded row stride (elements) for x/h activation tiles
#define WS        136         // padded row stride (elements) for transposed weights
#define NTHREADS  256         // 8 waves; wave w owns output columns [16w,16w+16)

// LDS layout (bf16 elements unless noted):
//   wt : 6 * 128 * WS      (Wr,Wz,Wh,Ur,Uz,Uh  transposed: [n][k])
//   xb : 2 * BQ * AS       (double-buffered x_t tile, row-major [m][k])
//   hb : 2 * BQ * AS       (double-buffered h tile,  row-major [m][k])
//   ab : 2 * BQ  floats    (double-buffered attention gate a_t)
#define WT_ELEMS (6 * 128 * WS)
#define XB_ELEMS (2 * BQ * AS)
#define HB_ELEMS (2 * BQ * AS)
#define LDS_BYTES ((WT_ELEMS + XB_ELEMS + HB_ELEMS) * 2 + 2 * BQ * 4)

__device__ __forceinline__ float fast_sigmoid(float x) {
  return __builtin_amdgcn_rcpf(1.0f + __expf(-x));
}
__device__ __forceinline__ float fast_tanh(float x) {
  float e = __expf(-2.0f * __builtin_fabsf(x));
  float t = (1.0f - e) * __builtin_amdgcn_rcpf(1.0f + e);
  return __builtin_copysignf(t, x);
}

// Workgroup barrier that only fences LDS (DScnt), leaving global loads/stores
// in flight across timesteps (split barrier, CDNA5 fine-grained counters).
__device__ __forceinline__ void wg_barrier_lds_only() {
  asm volatile("s_wait_dscnt 0x0" ::: "memory");
  asm volatile("s_barrier_signal -1" ::: "memory");
  asm volatile("s_barrier_wait -1" ::: "memory");
}

__device__ __forceinline__ v16bf cat16(v8bf lo, v8bf hi) {
  v16bf r;
#pragma unroll
  for (int i = 0; i < 8; ++i) { r[i] = lo[i]; r[i + 8] = hi[i]; }
  return r;
}

// A-operand fragment (16x32 bf16, M x K), source row-major [m][k], stride AS.
__device__ __forceinline__ v16bf load_a_frag(const __bf16* base, int lane, int kb) {
  const int m = lane & 15, half = lane >> 4;
  const __bf16* p = base + m * AS + kb + 8 * half;
  v8bf lo = *(const v8bf*)p;
  v8bf hi = *(const v8bf*)(p + 16);
  return cat16(lo, hi);
}

// B-operand fragment (32x16 bf16, K x N), weights stored transposed [n][k], stride WS.
__device__ __forceinline__ v16bf load_b_frag(const __bf16* wmat, int lane, int n0, int kb) {
  const int n = n0 + (lane & 15), half = lane >> 4;
  const __bf16* p = wmat + n * WS + kb + 8 * half;
  v8bf lo = *(const v8bf*)p;
  v8bf hi = *(const v8bf*)(p + 16);
  return cat16(lo, hi);
}

__device__ __forceinline__ v8f wmma_bf16(v16bf a, v16bf b, v8f c) {
  return __builtin_amdgcn_wmma_f32_16x16x32_bf16(false, a, false, b, (short)0, c,
                                                 false, false);
}

// Stage one 128x128 f32 weight matrix from global into LDS as transposed bf16.
__device__ __forceinline__ void stage_weight(const float* __restrict__ src,
                                             __bf16* dst, int tid) {
  for (int idx = tid; idx < UNITS * UNITS; idx += NTHREADS) {
    int k = idx >> 7, n = idx & 127;           // src is row-major W[k][n]
    dst[n * WS + k] = (__bf16)src[idx];
  }
}

__global__ __launch_bounds__(NTHREADS, 1)
void gru_scan_kernel(const float* __restrict__ inputs, const float* __restrict__ h0,
                     const float* __restrict__ Wr, const float* __restrict__ Ur,
                     const float* __restrict__ br,
                     const float* __restrict__ Wz, const float* __restrict__ Uz,
                     const float* __restrict__ bz,
                     const float* __restrict__ Wh, const float* __restrict__ Uh,
                     const float* __restrict__ bh,
                     float* __restrict__ out, int T) {
  extern __shared__ char smem[];
  __bf16* wt = (__bf16*)smem;                       // 6 transposed weight matrices
  __bf16* xb = wt + WT_ELEMS;                       // x tiles (x2)
  __bf16* hb = xb + XB_ELEMS;                       // h tiles (x2)
  float*  ab = (float*)(hb + HB_ELEMS);             // a values (x2)

  const int tid  = threadIdx.x;
  const int lane = tid & 31;
  const int wave = tid >> 5;
  const int b0   = blockIdx.x * BQ;

  // ---- stage weights (once) ----
  stage_weight(Wr, wt + 0 * 128 * WS, tid);
  stage_weight(Wz, wt + 1 * 128 * WS, tid);
  stage_weight(Wh, wt + 2 * 128 * WS, tid);
  stage_weight(Ur, wt + 3 * 128 * WS, tid);
  stage_weight(Uz, wt + 4 * 128 * WS, tid);
  stage_weight(Uh, wt + 5 * 128 * WS, tid);

  // ---- per-wave constants ----
  const int n0   = wave * 16;
  const int nc   = n0 + (lane & 15);   // this lane's output column
  const int half = lane >> 4;
  const float brv = br[nc], bzv = bz[nc], bhv = bh[nc];

  // ---- per-thread x-pipeline: thread owns row xm, cols xc0..xc0+7 ----
  const size_t strideT = (size_t)T * (UNITS + 1);   // per-batch-row stride in inputs
  const float* xbase = inputs + (size_t)b0 * strideT;
  const int xm  = tid >> 4;            // 0..15
  const int xc0 = (tid & 15) * 8;      // 0,8,...,120
  const float* xrowp = xbase + (size_t)xm * strideT + xc0;   // advances 129/step
  const bool  loadsA = (tid < BQ);     // threads 0..15 own a-column (c=128)
  const float* xap   = xbase + (size_t)tid * strideT + UNITS;

  // preload x for t = 0 into registers
  float xv[8];
#pragma unroll
  for (int j = 0; j < 8; ++j) xv[j] = xrowp[j];
  float aval = loadsA ? xap[0] : 0.0f;

  // ---- init h fragment from h0 (C-layout: lane=col nc, elem i -> row i+8*half)
  v8f h;
#pragma unroll
  for (int i = 0; i < 8; ++i) {
    const int m = i + 8 * half;
    float v = h0[(size_t)(b0 + m) * UNITS + nc];
    h[i] = v;
    hb[0 * BQ * AS + m * AS + nc] = (__bf16)v;      // h buffer 0 holds h0
  }

  // per-lane output base (rows 8*half..8*half+7, column nc), NT stores
  float* outRow = out + ((size_t)(b0 + 8 * half) * T) * UNITS + nc;

  for (int t = 0; t < T; ++t) {
    const int xpar = t & 1;
    const int hrd  = t & 1;        // read h_t from this buffer
    const int hwr  = hrd ^ 1;      // write h_{t+1} here

    // ---- spill register-staged x_t (loaded last iteration) into LDS ----
    {
      v8bf pk;
#pragma unroll
      for (int j = 0; j < 8; ++j) pk[j] = (__bf16)xv[j];
      *(v8bf*)(xb + xpar * BQ * AS + xm * AS + xc0) = pk;   // one ds_store_b128
      if (loadsA) ab[xpar * BQ + tid] = aval;
    }
    wg_barrier_lds_only();   // x_t/h_t visible; only DScnt fenced

    // ---- issue global loads for x_{t+1}; latency hidden by this step's math
    if (t + 1 < T) {
      const float* p = xrowp + (size_t)(t + 1) * (UNITS + 1);
#pragma unroll
      for (int j = 0; j < 8; ++j) xv[j] = p[j];
      if (loadsA) aval = xap[(size_t)(t + 1) * (UNITS + 1)];
      if (t + 2 < T)   // pull t+2 stream toward L2/WGP$ (global_prefetch)
        __builtin_prefetch(xrowp + (size_t)(t + 2) * (UNITS + 1), 0, 0);
    }

    // ---- 6 fused GEMM accumulations (biases folded into elementwise) ----
    v8f accr = {}, accz = {}, accxh = {}, acchh = {};
    const __bf16* xa = xb + xpar * BQ * AS;
    const __bf16* ha = hb + hrd * BQ * AS;
#pragma unroll
    for (int kb = 0; kb < UNITS; kb += 32) {
      v16bf aX = load_a_frag(xa, lane, kb);
      v16bf aH = load_a_frag(ha, lane, kb);
      accr  = wmma_bf16(aX, load_b_frag(wt + 0 * 128 * WS, lane, n0, kb), accr);
      accz  = wmma_bf16(aX, load_b_frag(wt + 1 * 128 * WS, lane, n0, kb), accz);
      accxh = wmma_bf16(aX, load_b_frag(wt + 2 * 128 * WS, lane, n0, kb), accxh);
      accr  = wmma_bf16(aH, load_b_frag(wt + 3 * 128 * WS, lane, n0, kb), accr);
      accz  = wmma_bf16(aH, load_b_frag(wt + 4 * 128 * WS, lane, n0, kb), accz);
      acchh = wmma_bf16(aH, load_b_frag(wt + 5 * 128 * WS, lane, n0, kb), acchh);
    }

    // ---- elementwise GRU update in accumulator layout ----
    __bf16* hw = hb + hwr * BQ * AS;
    v8f av = *(const v8f*)(ab + xpar * BQ + 8 * half);   // a[m] for this lane's rows
#pragma unroll
    for (int i = 0; i < 8; ++i) {
      const int m = i + 8 * half;
      float r   = fast_sigmoid(accr[i] + brv);
      float u   = fast_sigmoid(accz[i] + bzv);
      float htl = fast_tanh(accxh[i] + bhv + r * acchh[i]);
      float uh  = av[i] * u;
      float hn  = h[i] + uh * (htl - h[i]);   // (1-uh)*h + uh*htilde
      h[i] = hn;
      hw[m * AS + nc] = (__bf16)hn;           // A-operand for next step
      __builtin_nontemporal_store(hn, outRow + (size_t)i * T * UNITS +
                                          (size_t)t * UNITS);
    }
  }
}

extern "C" void kernel_launch(void* const* d_in, const int* in_sizes, int n_in,
                              void* d_out, int out_size, void* d_ws, size_t ws_size,
                              hipStream_t stream) {
  const float* inputs = (const float*)d_in[0];
  const float* h0     = (const float*)d_in[1];
  const float* W_r    = (const float*)d_in[2];
  const float* U_r    = (const float*)d_in[3];
  const float* b_r    = (const float*)d_in[4];
  const float* W_z    = (const float*)d_in[5];
  const float* U_z    = (const float*)d_in[6];
  const float* b_z    = (const float*)d_in[7];
  const float* W_h    = (const float*)d_in[8];
  const float* U_h    = (const float*)d_in[9];
  const float* b_h    = (const float*)d_in[10];
  float* out = (float*)d_out;

  const int B = in_sizes[1] / UNITS;                 // 2048
  const int T = in_sizes[0] / (B * (UNITS + 1));     // 200

  (void)hipFuncSetAttribute(reinterpret_cast<const void*>(gru_scan_kernel),
                            hipFuncAttributeMaxDynamicSharedMemorySize, LDS_BYTES);

  dim3 grid(B / BQ);      // 128 blocks
  dim3 block(NTHREADS);   // 8 waves
  gru_scan_kernel<<<grid, block, LDS_BYTES, stream>>>(
      inputs, h0, W_r, U_r, b_r, W_z, U_z, b_z, W_h, U_h, b_h, out, T);
}
